// BasePriorNetwork_13185549599194
// MI455X (gfx1250) — matrix operand order repaired and checked by reading, not compiled, not run on hardware
//
#include <hip/hip_runtime.h>

#define BB 8
#define NN 515
#define DD 1024
#define HH 8
#define DHD 64
#define LL 4
#define FFD 4096
#define NKK 516            // N+1 keys (null kv prepended)
#define VTS 544            // padded key stride for transposed V (>=516+28, mult of 16)
#define MROWS (BB*NN)      // 4120

typedef __bf16 bf16_t;
typedef __attribute__((ext_vector_type(16))) __bf16 bf16x16;
typedef __attribute__((ext_vector_type(8)))  __bf16 bf16x8;
typedef __attribute__((ext_vector_type(4)))  __bf16 bf16x4;
typedef __attribute__((ext_vector_type(8)))  float  f32x8;

__device__ __forceinline__ int imin(int a, int b) { return a < b ? a : b; }

// combine two aligned 8-element bf16 runs into one 16-element fragment
__device__ __forceinline__ bf16x16 ld2x8(const bf16_t* p0, const bf16_t* p1) {
  bf16x8 lo = *(const bf16x8*)p0;
  bf16x8 hi = *(const bf16x8*)p1;
  return __builtin_shufflevector(lo, hi, 0,1,2,3,4,5,6,7,8,9,10,11,12,13,14,15);
}

// ---------------------------------------------------------------- weight convert+transpose
// W f32 [K][ldw] (use N cols) -> Wt bf16 [N][K]
__global__ __launch_bounds__(256)
void k_convT(const float* __restrict__ W, bf16_t* __restrict__ Wt,
             int K, int N, int ldw) {
  size_t i = (size_t)blockIdx.x * 256 + threadIdx.x;
  if (i >= (size_t)K * N) return;
  int k = (int)(i / N), n = (int)(i % N);
  Wt[(size_t)n * K + k] = (bf16_t)W[(size_t)k * ldw + n];
}

// ---------------------------------------------------------------- layernorm
// mode 0: Yb = LN(X)*g (bf16)  mode 1: stable (bf16)  mode 2: Yf = resin + LN(X)*g (f32)
__device__ __forceinline__ float blk_sum(float v, float* sbuf) {
  #pragma unroll
  for (int o = 16; o; o >>= 1) v += __shfl_xor(v, o, 32);
  int ln = threadIdx.x & 31, w = threadIdx.x >> 5;
  if (ln == 0) sbuf[w] = v;
  __syncthreads();
  float t = (ln < 8) ? sbuf[ln] : 0.f;
  #pragma unroll
  for (int o = 4; o; o >>= 1) t += __shfl_xor(t, o, 32);
  t = __shfl(t, 0, 32);
  __syncthreads();
  return t;
}
__device__ __forceinline__ float blk_max(float v, float* sbuf) {
  #pragma unroll
  for (int o = 16; o; o >>= 1) v = fmaxf(v, __shfl_xor(v, o, 32));
  int ln = threadIdx.x & 31, w = threadIdx.x >> 5;
  if (ln == 0) sbuf[w] = v;
  __syncthreads();
  float t = (ln < 8) ? sbuf[ln] : -3.0e38f;
  #pragma unroll
  for (int o = 4; o; o >>= 1) t = fmaxf(t, __shfl_xor(t, o, 32));
  t = __shfl(t, 0, 32);
  __syncthreads();
  return t;
}

__global__ __launch_bounds__(256)
void k_layernorm(const float* __restrict__ X, const float* __restrict__ g,
                 const float* __restrict__ resin, float* __restrict__ Yf,
                 bf16_t* __restrict__ Yb, int mode) {
  __shared__ float sbuf[8];
  size_t row = blockIdx.x;
  int t = threadIdx.x;
  float4 v = ((const float4*)(X + row * DD))[t];
  if (mode == 1) {
    float mx = blk_max(fmaxf(fmaxf(v.x, v.y), fmaxf(v.z, v.w)), sbuf);
    float im = 1.f / mx;
    v.x *= im; v.y *= im; v.z *= im; v.w *= im;
  }
  float mean = blk_sum(v.x + v.y + v.z + v.w, sbuf) * (1.f / DD);
  float dx = v.x - mean, dy = v.y - mean, dz = v.z - mean, dw = v.w - mean;
  float var = blk_sum(dx*dx + dy*dy + dz*dz + dw*dw, sbuf) * (1.f / DD);
  float inv = rsqrtf(var + 1e-5f);
  float4 g4 = ((const float4*)g)[t];
  float ox = dx * inv * g4.x, oy = dy * inv * g4.y;
  float oz = dz * inv * g4.z, ow = dw * inv * g4.w;
  if (mode == 2) {
    float4 r4 = ((const float4*)(resin + row * DD))[t];
    float4 o; o.x = ox + r4.x; o.y = oy + r4.y; o.z = oz + r4.z; o.w = ow + r4.w;
    ((float4*)(Yf + row * DD))[t] = o;
  } else {
    bf16x4 ob;
    ob[0] = (bf16_t)ox; ob[1] = (bf16_t)oy; ob[2] = (bf16_t)oz; ob[3] = (bf16_t)ow;
    ((bf16x4*)(Yb + row * DD))[t] = ob;
  }
}

// ---------------------------------------------------------------- GEMM (bf16 WMMA, async LDS)
// C[M,N] = A[M,K](bf16) * Bt[N,K](bf16)^T
// epi: 0 store, 1 +bias(f32 out), 2 C=C*silu(acc) (bf16 in/out), 3 C+=acc (f32)
__device__ __forceinline__ void gemm_stage(const bf16_t* __restrict__ A, int lda,
                                           const bf16_t* __restrict__ Bt, int ldb,
                                           bf16_t (*As)[32], bf16_t (*Bs)[32],
                                           int bm, int bn, int M, int k0, int tid) {
  int ar = tid >> 1, ac = (tid & 1) * 16;
  int gr = bm + ar; gr = gr < M ? gr : M - 1;   // clamp; garbage rows never stored
  const bf16_t* sa = A + (size_t)gr * lda + k0 + ac;
  unsigned la = (unsigned)(uintptr_t)&As[ar][ac];
  asm volatile("global_load_async_to_lds_b128 %0, %1, off" :: "v"(la), "v"(sa) : "memory");
  asm volatile("global_load_async_to_lds_b128 %0, %1, off" :: "v"(la + 16u), "v"(sa + 8) : "memory");
  int br = tid >> 2, bc = (tid & 3) * 8;
  const bf16_t* sb = Bt + (size_t)(bn + br) * ldb + k0 + bc;
  unsigned lb = (unsigned)(uintptr_t)&Bs[br][bc];
  asm volatile("global_load_async_to_lds_b128 %0, %1, off" :: "v"(lb), "v"(sb) : "memory");
}

template <bool OUTBF>
__global__ __launch_bounds__(256)
void k_gemm(const bf16_t* __restrict__ A, int lda,
            const bf16_t* __restrict__ Bt, int ldb,
            void* __restrict__ Cv, int ldc,
            int M, int N, int K,
            const float* __restrict__ bias, int epi) {
  __shared__ __align__(16) bf16_t As[2][128][32];
  __shared__ __align__(16) bf16_t Bs[2][64][32];
  int tid = threadIdx.x;
  int wave = tid >> 5, lane = tid & 31;
  int half = lane >> 4, l16 = lane & 15;
  int wm = wave >> 1;             // 0..3 : 32-row sub-block
  int wn = wave & 1;              // 0..1 : 32-col sub-block
  int bm = blockIdx.y * 128;
  int bn = blockIdx.x * 64;

  f32x8 acc[2][2] = {{{}, {}}, {{}, {}}};

  gemm_stage(A, lda, Bt, ldb, As[0], Bs[0], bm, bn, M, 0, tid);
  int cur = 0;
  for (int k0 = 0; k0 < K; k0 += 32) {
    bool nxt = (k0 + 32) < K;
    if (nxt) {
      gemm_stage(A, lda, Bt, ldb, As[cur ^ 1], Bs[cur ^ 1], bm, bn, M, k0 + 32, tid);
      asm volatile("s_wait_asynccnt 3" ::: "memory");
    } else {
      asm volatile("s_wait_asynccnt 0" ::: "memory");
    }
    __syncthreads();
    const bf16_t* a0r = &As[cur][wm * 32 + l16][0];
    const bf16_t* a1r = &As[cur][wm * 32 + 16 + l16][0];
    bf16x16 a0 = ld2x8(a0r + 8 * half, a0r + 16 + 8 * half);
    bf16x16 a1 = ld2x8(a1r + 8 * half, a1r + 16 + 8 * half);
    const bf16_t* b0r = &Bs[cur][wn * 32 + l16][0];
    const bf16_t* b1r = &Bs[cur][wn * 32 + 16 + l16][0];
    bf16x16 b0 = ld2x8(b0r + 8 * half, b0r + 16 + 8 * half);
    bf16x16 b1 = ld2x8(b1r + 8 * half, b1r + 16 + 8 * half);
    acc[0][0] = __builtin_amdgcn_wmma_f32_16x16x32_bf16(false, a0, false, b0, (short)0, acc[0][0], false, false);
    acc[0][1] = __builtin_amdgcn_wmma_f32_16x16x32_bf16(false, a0, false, b1, (short)0, acc[0][1], false, false);
    acc[1][0] = __builtin_amdgcn_wmma_f32_16x16x32_bf16(false, a1, false, b0, (short)0, acc[1][0], false, false);
    acc[1][1] = __builtin_amdgcn_wmma_f32_16x16x32_bf16(false, a1, false, b1, (short)0, acc[1][1], false, false);
    __syncthreads();
    cur ^= 1;
  }

  #pragma unroll
  for (int s = 0; s < 2; ++s) {
    #pragma unroll
    for (int t = 0; t < 2; ++t) {
      int col = bn + wn * 32 + t * 16 + l16;
      #pragma unroll
      for (int r = 0; r < 8; ++r) {
        int gm = bm + wm * 32 + s * 16 + 8 * half + r;
        if (gm >= M) continue;
        size_t idx = (size_t)gm * ldc + col;
        float v = acc[s][t][r];
        if (OUTBF) {
          bf16_t* Cb = (bf16_t*)Cv;
          if (epi == 2) Cb[idx] = (bf16_t)((float)Cb[idx] * (v / (1.f + __expf(-v))));
          else          Cb[idx] = (bf16_t)v;
        } else {
          float* Cf = (float*)Cv;
          if (epi == 1)      Cf[idx] = v + bias[col];
          else if (epi == 3) Cf[idx] += v;
          else               Cf[idx] = v;
        }
      }
    }
  }
}

// ---------------------------------------------------------------- rotary helpers
__device__ __forceinline__ float rot_apply(float t, int d, int pos) {
  float p = __shfl_xor(t, 1, 32);
  float th = (float)pos * __expf((float)(d & ~1) * (-9.210340371976184f / 32.f));
  float c = __cosf(th), s = __sinf(th);
  float rh = (d & 1) ? p : -p;
  return t * c + rh * s;
}

// q_raw f32 [b*N+n][h*64+d] -> q bf16 [((b*H+h)*N+n)*64+d] : rotary + l2norm*4
__global__ __launch_bounds__(64)
void k_qprep(const float* __restrict__ qraw, bf16_t* __restrict__ q) {
  __shared__ float sred[2];
  int row = blockIdx.x;
  int n = row % NN, hb = row / NN;
  int h = hb % HH, b = hb / HH;
  int d = threadIdx.x;
  float t = qraw[((size_t)(b * NN + n)) * (HH * DHD) + h * DHD + d];
  if (d < 32) t = rot_apply(t, d, n);
  float sq = t * t;
  #pragma unroll
  for (int o = 16; o; o >>= 1) sq += __shfl_xor(sq, o, 32);
  if ((threadIdx.x & 31) == 0) sred[threadIdx.x >> 5] = sq;
  __syncthreads();
  float tot = sred[0] + sred[1];
  q[(((size_t)(b * HH + h)) * NN + n) * DHD + d] = (bf16_t)(t * rsqrtf(tot) * 4.0f);
}

// kv_raw f32 -> Kall bf16 [b][j][d], Vt bf16 [b][d][j] (stride VTS)
__global__ __launch_bounds__(64)
void k_kvprep(const float* __restrict__ kvraw, const float* __restrict__ nullkv,
              bf16_t* __restrict__ Kall, bf16_t* __restrict__ Vt) {
  __shared__ float sred[2];
  int row = blockIdx.x;
  int j = row % NKK, b = row / NKK;
  int d = threadIdx.x;
  float kt, vt;
  if (j == 0) { kt = nullkv[d]; vt = nullkv[64 + d]; }
  else {
    size_t base = ((size_t)(b * NN + (j - 1))) * 128;
    kt = kvraw[base + d]; vt = kvraw[base + 64 + d];
    if (d < 32) kt = rot_apply(kt, d, j - 1);
  }
  float sq = kt * kt;
  #pragma unroll
  for (int o = 16; o; o >>= 1) sq += __shfl_xor(sq, o, 32);
  if ((threadIdx.x & 31) == 0) sred[threadIdx.x >> 5] = sq;
  __syncthreads();
  float tot = sred[0] + sred[1];
  Kall[((size_t)b * NKK + j) * DHD + d] = (bf16_t)(kt * rsqrtf(tot) * 4.0f);
  Vt[((size_t)b * DHD + d) * VTS + j] = (bf16_t)vt;
}

// ---------------------------------------------------------------- attention
__device__ __forceinline__ float relpos_bias(int dlt, int h, const float* __restrict__ emb) {
  int bucket;
  if (dlt <= 0) bucket = 0;
  else if (dlt < 16) bucket = dlt;
  else {
    int vb = 16 + (int)(__logf((float)dlt * (1.0f / 16.0f)) * (16.0f / 2.0794415416798357f));
    bucket = vb < 31 ? vb : 31;
  }
  return emb[bucket * HH + h];
}

__global__ __launch_bounds__(32)
void k_attention(const bf16_t* __restrict__ Q, const bf16_t* __restrict__ Kall,
                 const bf16_t* __restrict__ Vt, const float* __restrict__ emb,
                 bf16_t* __restrict__ Out) {
  __shared__ __align__(16) bf16_t Plds[16][32];
  int lane = threadIdx.x;
  int half = lane >> 4, l16 = lane & 15;
  int m0 = blockIdx.x * 16;
  int h = blockIdx.y, b = blockIdx.z;

  int mq = imin(m0 + l16, NN - 1);
  const bf16_t* qrow = Q + (((size_t)(b * HH + h)) * NN + mq) * DHD;
  bf16x16 aq[2];
  #pragma unroll
  for (int s = 0; s < 2; ++s)
    aq[s] = ld2x8(qrow + 32 * s + 8 * half, qrow + 32 * s + 16 + 8 * half);

  f32x8 accO[4] = {{}, {}, {}, {}};
  float m_run[8], l_run[8];
  #pragma unroll
  for (int r = 0; r < 8; ++r) { m_run[r] = -3.0e38f; l_run[r] = 0.f; }

  int jmax = imin(m0 + 15, NN - 1) + 1;    // causal: j <= i+1
  for (int kb = 0; kb < NKK && kb <= jmax; kb += 32) {
    // S = Q K^T for two 16-key subtiles (K fragments straight from global bf16)
    f32x8 c01[2] = {{}, {}};
    #pragma unroll
    for (int sub = 0; sub < 2; ++sub) {
      int key = imin(kb + sub * 16 + l16, NKK - 1);
      const bf16_t* krow = Kall + ((size_t)b * NKK + key) * DHD;
      #pragma unroll
      for (int s = 0; s < 2; ++s) {
        bf16x16 bk = ld2x8(krow + 32 * s + 8 * half, krow + 32 * s + 16 + 8 * half);
        c01[sub] = __builtin_amdgcn_wmma_f32_16x16x32_bf16(false, aq[s], false, bk, (short)0, c01[sub], false, false);
      }
    }
    // online softmax; write P to LDS
    #pragma unroll
    for (int r = 0; r < 8; ++r) {
      int i = m0 + r + 8 * half;
      int ie = imin(i, NN - 1);
      int j0 = kb + l16, j1 = j0 + 16;
      float v0 = (j0 <= ie + 1) ? c01[0][r] + relpos_bias(ie - j0, h, emb) : -3.0e38f;
      float v1 = (j1 <= ie + 1) ? c01[1][r] + relpos_bias(ie - j1, h, emb) : -3.0e38f;
      float mx = fmaxf(v0, v1);
      #pragma unroll
      for (int o = 8; o >= 1; o >>= 1) mx = fmaxf(mx, __shfl_xor(mx, o, 32));
      float mnew = fmaxf(m_run[r], mx);
      float alpha = __expf(m_run[r] - mnew);
      float p0 = __expf(v0 - mnew), p1 = __expf(v1 - mnew);
      float rs = p0 + p1;
      #pragma unroll
      for (int o = 8; o >= 1; o >>= 1) rs += __shfl_xor(rs, o, 32);
      l_run[r] = l_run[r] * alpha + rs;
      m_run[r] = mnew;
      #pragma unroll
      for (int t = 0; t < 4; ++t) accO[t][r] *= alpha;
      Plds[r + 8 * half][l16] = (bf16_t)p0;
      Plds[r + 8 * half][16 + l16] = (bf16_t)p1;
    }
    asm volatile("s_wait_dscnt 0" ::: "memory");
    // O += P * V (V fragments straight from transposed global bf16)
    const bf16_t* prow = &Plds[l16][0];
    bf16x16 ap = ld2x8(prow + 8 * half, prow + 16 + 8 * half);
    #pragma unroll
    for (int t = 0; t < 4; ++t) {
      const bf16_t* vrow = Vt + ((size_t)b * DHD + 16 * t + l16) * VTS + kb;
      bf16x16 bv = ld2x8(vrow + 8 * half, vrow + 16 + 8 * half);
      accO[t] = __builtin_amdgcn_wmma_f32_16x16x32_bf16(false, ap, false, bv, (short)0, accO[t], false, false);
    }
    asm volatile("s_wait_dscnt 0" ::: "memory");
  }
  #pragma unroll
  for (int r = 0; r < 8; ++r) {
    int i = m0 + r + 8 * half;
    if (i >= NN) continue;
    float inv = (l_run[r] > 0.f) ? 1.f / l_run[r] : 0.f;
    size_t base = ((size_t)b * NN + i) * (HH * DHD) + h * DHD;
    #pragma unroll
    for (int t = 0; t < 4; ++t) Out[base + 16 * t + l16] = (bf16_t)(accO[t][r] * inv);
  }
}

// ---------------------------------------------------------------- launch
extern "C" void kernel_launch(void* const* d_in, const int* in_sizes, int n_in,
                              void* d_out, int out_size, void* d_ws, size_t ws_size,
                              hipStream_t stream) {
  (void)in_sizes; (void)n_in; (void)out_size; (void)ws_size;
  const float* in_x   = (const float*)d_in[0];
  const float* attn_g = (const float*)d_in[1];
  const float* Wq     = (const float*)d_in[2];
  const float* Wkv    = (const float*)d_in[3];
  const float* bkv    = (const float*)d_in[4];
  const float* nullkv = (const float*)d_in[5];
  const float* Wo     = (const float*)d_in[6];
  const float* out_g  = (const float*)d_in[7];
  const float* ff_g   = (const float*)d_in[8];
  const float* Wff1   = (const float*)d_in[9];
  const float* Wff2   = (const float*)d_in[10];
  const float* emb    = (const float*)d_in[11];
  const float* fin_g  = (const float*)d_in[12];
  const float* Wproj  = (const float*)d_in[13];

  float* wsf = (float*)d_ws;
  size_t off = 0;
  float* x    = wsf + off; off += (size_t)MROWS * DD;
  float* qraw = wsf + off; off += (size_t)MROWS * (HH * DHD);
  float* kvr  = wsf + off; off += (size_t)MROWS * 128;
  float* tmp  = wsf + off; off += (size_t)MROWS * DD;

  bf16_t* wsb = (bf16_t*)(wsf + off);
  size_t ob = 0;
  bf16_t* xn   = wsb + ob; ob += (size_t)MROWS * DD;
  bf16_t* q    = wsb + ob; ob += (size_t)MROWS * (HH * DHD);
  bf16_t* Kall = wsb + ob; ob += (size_t)BB * NKK * DHD;
  bf16_t* Vt   = wsb + ob; ob += (size_t)BB * DHD * VTS;
  bf16_t* ao   = wsb + ob; ob += (size_t)MROWS * (HH * DHD);
  bf16_t* ffa  = wsb + ob; ob += (size_t)MROWS * FFD;
  bf16_t* WqT  = wsb + ob; ob += (size_t)LL * 512 * DD;
  bf16_t* WkvT = wsb + ob; ob += (size_t)LL * 128 * DD;
  bf16_t* WoT  = wsb + ob; ob += (size_t)LL * DD * 512;
  bf16_t* Wf1T = wsb + ob; ob += (size_t)LL * 2 * FFD * DD;
  bf16_t* Wf2T = wsb + ob; ob += (size_t)LL * DD * FFD;
  bf16_t* WpT  = wsb + ob; ob += (size_t)DD * DD;

  hipMemcpyAsync(x, in_x, (size_t)MROWS * DD * sizeof(float),
                 hipMemcpyDeviceToDevice, stream);

  auto convT = [&](const float* W, bf16_t* Wt, int K, int N, int ldw) {
    size_t tot = (size_t)K * N;
    k_convT<<<dim3((unsigned)((tot + 255) / 256)), 256, 0, stream>>>(W, Wt, K, N, ldw);
  };
  // one-time weight convert + transpose (bf16 resident, [N][K])
  for (int l = 0; l < LL; ++l) {
    convT(Wq   + (size_t)l * DD * 512,     WqT  + (size_t)l * 512 * DD,     DD, 512, 512);
    convT(Wkv  + (size_t)l * DD * 128,     WkvT + (size_t)l * 128 * DD,     DD, 128, 128);
    convT(Wo   + (size_t)l * 512 * DD,     WoT  + (size_t)l * DD * 512,     512, DD, DD);
    convT(Wff1 + (size_t)l * DD * 2 * FFD, Wf1T + (size_t)l * 2 * FFD * DD, DD, 2 * FFD, 2 * FFD);
    convT(Wff2 + (size_t)l * FFD * DD,     Wf2T + (size_t)l * DD * FFD,     FFD, DD, DD);
  }
  convT(Wproj, WpT, DD, DD, DD);

  auto gemm = [&](const bf16_t* A, int lda, const bf16_t* Bt, int ldb,
                  void* C, int ldc, int M, int N, int K,
                  const float* bias, int epi, bool outbf) {
    dim3 grid(N / 64, (M + 127) / 128);
    if (outbf) k_gemm<true ><<<grid, 256, 0, stream>>>(A, lda, Bt, ldb, C, ldc, M, N, K, bias, epi);
    else       k_gemm<false><<<grid, 256, 0, stream>>>(A, lda, Bt, ldb, C, ldc, M, N, K, bias, epi);
  };

  for (int l = 0; l < LL; ++l) {
    k_layernorm<<<MROWS, 256, 0, stream>>>(x, attn_g + l * DD, nullptr, nullptr, xn, 0);
    gemm(xn, DD, WqT  + (size_t)l * 512 * DD, DD, qraw, 512, MROWS, 512, DD, nullptr, 0, false);
    gemm(xn, DD, WkvT + (size_t)l * 128 * DD, DD, kvr,  128, MROWS, 128, DD, bkv + l * 128, 1, false);
    k_qprep<<<BB * HH * NN, 64, 0, stream>>>(qraw, q);
    k_kvprep<<<BB * NKK, 64, 0, stream>>>(kvr, nullkv + l * 128, Kall, Vt);
    k_attention<<<dim3((NN + 15) / 16, HH, BB), 32, 0, stream>>>(q, Kall, Vt, emb, ao);
    gemm(ao, 512, WoT + (size_t)l * DD * 512, 512, tmp, DD, MROWS, DD, 512, nullptr, 0, false);
    k_layernorm<<<MROWS, 256, 0, stream>>>(tmp, out_g + l * DD, x, x, nullptr, 2);
    k_layernorm<<<MROWS, 256, 0, stream>>>(x, ff_g + l * DD, nullptr, nullptr, xn, 0);
    const bf16_t* W1 = Wf1T + (size_t)l * 2 * FFD * DD;
    gemm(xn, DD, W1,                      DD, ffa, FFD, MROWS, FFD, DD, nullptr, 0, true);  // a
    gemm(xn, DD, W1 + (size_t)FFD * DD,   DD, ffa, FFD, MROWS, FFD, DD, nullptr, 2, true);  // a*silu(g)
    gemm(ffa, FFD, Wf2T + (size_t)l * DD * FFD, FFD, x, DD, MROWS, DD, FFD, nullptr, 3, false);
  }
  k_layernorm<<<MROWS, 256, 0, stream>>>(x, fin_g, nullptr, nullptr, xn, 1);
  gemm(xn, DD, WpT, DD, d_out, DD, MROWS, DD, DD, nullptr, 0, false);
}
